// DualGCN_36636071035178
// MI455X (gfx1250) — compile-verified, matrix operand order checked by reading, not run on hardware
//
#include <hip/hip_runtime.h>
#include <cstdint>

// ---------------------------------------------------------------------------
// DualGCN on MI455X (gfx1250, wave32).
// Dense GEMMs -> v_wmma_f32_16x16x32_bf16 (f32 accumulate), 2x2 register
// tiling per wave (4 WMMAs per 2 A + 2 B fragment loads -> ~16 FLOP/byte).
// Graph aggregation -> float4 gathers + global_atomic_add_f32 (L2-resident).
// ---------------------------------------------------------------------------

static const int N_NODES = 20000;
static const int N_EDGES = 320000;
static const int IN_C    = 2000;
static const int KP_RNA  = 2016;   // 2000 padded to multiple of 32
static const int HID     = 512;
static const int OUT_F   = 256;
static const int QDIM    = 100;
static const int KP_ADT  = 128;    // 100 padded to multiple of 32

typedef __attribute__((ext_vector_type(16))) __bf16 v16bf;
typedef __attribute__((ext_vector_type(8)))  float  v8f;

union BF16x16 { v16bf v; uint32_t u[8]; };

__device__ __forceinline__ uint16_t f2bf(float f) {
    uint32_t u = __float_as_uint(f);
    uint32_t r = u + 0x7FFFu + ((u >> 16) & 1u);   // round-to-nearest-even
    return (uint16_t)(r >> 16);
}

// ------------------------------ graph prep ---------------------------------

__global__ void k_fill1(float* p, int n) {
    int i = blockIdx.x * blockDim.x + threadIdx.x;
    if (i < n) p[i] = 1.0f;                         // self-loop weight
}

__global__ void k_scatter_deg(float* deg, const int* __restrict__ dst,
                              const float* __restrict__ w, int e) {
    int i = blockIdx.x * blockDim.x + threadIdx.x;
    if (i < e) atomicAdd(&deg[dst[i]], w[i]);
}

__global__ void k_to_dinv(float* d, int n) {
    int i = blockIdx.x * blockDim.x + threadIdx.x;
    if (i < n) { float v = d[i]; d[i] = (v > 0.0f) ? rsqrtf(v) : 0.0f; }
}

__global__ void k_edge_norm(float* __restrict__ norm, const float* __restrict__ dinv,
                            const int* __restrict__ src, const int* __restrict__ dst,
                            const float* __restrict__ w, int e) {
    int i = blockIdx.x * blockDim.x + threadIdx.x;
    if (i < e) norm[i] = dinv[src[i]] * w[i] * dinv[dst[i]];
}

// --------------------------- bf16 conversions ------------------------------

// dst[m*Kp + k] = k<K ? bf16(src[m*K + k]) : 0     (grid.y = M)
__global__ void k_cvt_pad_bf16(uint16_t* __restrict__ dst, const float* __restrict__ src,
                               int K, int Kp) {
    int k = blockIdx.x * blockDim.x + threadIdx.x;
    int m = blockIdx.y;
    if (k >= Kp) return;
    dst[(size_t)m * Kp + k] = (k < K) ? f2bf(src[(size_t)m * K + k]) : (uint16_t)0;
}

// dst[n*Kp + k] = k<K ? bf16(W[k*Ncols + n]) : 0   (grid.y = Ncols) -> W^T, padded
__global__ void k_cvt_wT_bf16(uint16_t* __restrict__ dst, const float* __restrict__ W,
                              int K, int Ncols, int Kp) {
    int k = blockIdx.x * blockDim.x + threadIdx.x;
    int n = blockIdx.y;
    if (k >= Kp) return;
    dst[(size_t)n * Kp + k] = (k < K) ? f2bf(W[(size_t)k * Ncols + n]) : (uint16_t)0;
}

// dst[m*ldDst + colOff + f] = bf16(src[m*F + f])   (concat packing), F = 1<<fsh
__global__ void k_pack_bf16(uint16_t* __restrict__ dst, const float* __restrict__ src,
                            int total, int ldDst, int colOff, int fsh) {
    int idx = blockIdx.x * blockDim.x + threadIdx.x;
    if (idx >= total) return;
    int m = idx >> fsh;
    int f = idx & ((1 << fsh) - 1);
    dst[(size_t)m * ldDst + colOff + f] = f2bf(src[idx]);
}

__global__ void k_relu_bf16(uint16_t* __restrict__ dst, const float* __restrict__ src,
                            int total) {
    int idx = blockIdx.x * blockDim.x + threadIdx.x;
    if (idx >= total) return;
    float v = src[idx];
    dst[idx] = f2bf(v > 0.0f ? v : 0.0f);
}

// ------------------------------ WMMA GEMM ----------------------------------
// C[M x Ncols] = A[M x Kp] * B[Kp x Ncols] (+bias), A row-major bf16,
// Bt = B^T row-major bf16 (Ncols x Kp). Each wave computes a 32x32 patch
// (2x2 tiles of 16x16): per k-step, 2 A + 2 B fragments feed 4 WMMAs.
// Fragment addressing follows the CDNA5 16-bit 16x32 layout: lanes 0-15 hold
// K={0..7,16..23}, lanes 16-31 hold K={8..15,24..31}, row/col = lane & 15.

#define GEMM_WAVES 4

__device__ __forceinline__ void load_frag(BF16x16& f, const uint16_t* row, int k, int kk) {
    const uint32_t* p = (const uint32_t*)(row + k);
#pragma unroll
    for (int i = 0; i < 4; ++i) {
        f.u[i]     = p[kk + i];
        f.u[4 + i] = p[kk + 8 + i];
    }
}

__global__ void k_gemm_bf16_wmma(const uint16_t* __restrict__ A,
                                 const uint16_t* __restrict__ Bt,
                                 float* __restrict__ C,
                                 const float* __restrict__ bias,
                                 int Ncols, int Kp) {
    const int lane  = threadIdx.x & 31;
    const int wave  = threadIdx.x >> 5;
    const int mBase = blockIdx.x * 32;                             // 2 M-tiles
    const int nBase = (blockIdx.y * GEMM_WAVES + wave) * 32;       // 2 N-tiles
    const int l15   = lane & 15;
    const int kk    = (lane >> 4) * 4;   // dword sub-offset (8 bf16) for upper half-wave

    const uint16_t* Arow0 = A  + (size_t)(mBase + l15)      * Kp;
    const uint16_t* Arow1 = A  + (size_t)(mBase + 16 + l15) * Kp;
    const uint16_t* Brow0 = Bt + (size_t)(nBase + l15)      * Kp;
    const uint16_t* Brow1 = Bt + (size_t)(nBase + 16 + l15) * Kp;

    v8f acc00 = {}, acc01 = {}, acc10 = {}, acc11 = {};
    for (int k = 0; k < Kp; k += 32) {
        BF16x16 fa0, fa1, fb0, fb1;
        load_frag(fa0, Arow0, k, kk);
        load_frag(fa1, Arow1, k, kk);
        load_frag(fb0, Brow0, k, kk);
        load_frag(fb1, Brow1, k, kk);
        acc00 = __builtin_amdgcn_wmma_f32_16x16x32_bf16(false, fa0.v, false, fb0.v,
                                                        (short)0, acc00, false, false);
        acc01 = __builtin_amdgcn_wmma_f32_16x16x32_bf16(false, fa0.v, false, fb1.v,
                                                        (short)0, acc01, false, false);
        acc10 = __builtin_amdgcn_wmma_f32_16x16x32_bf16(false, fa1.v, false, fb0.v,
                                                        (short)0, acc10, false, false);
        acc11 = __builtin_amdgcn_wmma_f32_16x16x32_bf16(false, fa1.v, false, fb1.v,
                                                        (short)0, acc11, false, false);
    }

    // Store: VGPR r of a 16x16 f32 tile holds row M=r (lanes 0-15) / M=8+r
    // (lanes 16-31), column N = lane & 15.
    const int rOff = (lane >> 4) * 8;
    const float bc0 = bias ? bias[nBase + l15]      : 0.0f;
    const float bc1 = bias ? bias[nBase + 16 + l15] : 0.0f;
    float* C00 = C + (size_t)(mBase + rOff)      * Ncols + (nBase + l15);
    float* C10 = C + (size_t)(mBase + 16 + rOff) * Ncols + (nBase + l15);
#pragma unroll
    for (int r = 0; r < 8; ++r) {
        C00[(size_t)r * Ncols]      = acc00[r] + bc0;
        C00[(size_t)r * Ncols + 16] = acc01[r] + bc1;
        C10[(size_t)r * Ncols]      = acc10[r] + bc0;
        C10[(size_t)r * Ncols + 16] = acc11[r] + bc1;
    }
}

// --------------------------- GCN aggregation -------------------------------

// out = h * dinv[i]^2 (self-loop, w=1) + bias[f] ;  F = 1<<fsh
__global__ void k_agg_init(float* __restrict__ out, const float* __restrict__ h,
                           const float* __restrict__ dinv, const float* __restrict__ bias,
                           int total, int fsh) {
    int idx = blockIdx.x * blockDim.x + threadIdx.x;
    if (idx >= total) return;
    int i = idx >> fsh;
    int f = idx & ((1 << fsh) - 1);
    float di = dinv[i];
    out[idx] = h[idx] * di * di + bias[f];
}

// out[dst[e]*F + f..f+3] += h[src[e]*F + f..f+3] * norm[e] ; total = E*(F/4)
__global__ void k_agg_scatter(float* __restrict__ out, const float* __restrict__ h,
                              const int* __restrict__ src, const int* __restrict__ dst,
                              const float* __restrict__ norm,
                              int total, int F, int fsh4) {
    int idx = blockIdx.x * blockDim.x + threadIdx.x;
    if (idx >= total) return;
    int e = idx >> fsh4;
    int f = (idx & ((1 << fsh4) - 1)) << 2;
    float s = norm[e];
    const float4 hv = *reinterpret_cast<const float4*>(h + (size_t)src[e] * F + f);
    float* o = out + (size_t)dst[e] * F + f;
    atomicAdd(o + 0, hv.x * s);
    atomicAdd(o + 1, hv.y * s);
    atomicAdd(o + 2, hv.z * s);
    atomicAdd(o + 3, hv.w * s);
}

// ---------------------------------------------------------------------------

extern "C" void kernel_launch(void* const* d_in, const int* in_sizes, int n_in,
                              void* d_out, int out_size, void* d_ws, size_t ws_size,
                              hipStream_t stream) {
    const float* x_RNA  = (const float*)d_in[0];
    const float* x_ADT  = (const float*)d_in[1];
    const int*   simEI  = (const int*)  d_in[2];
    const float* simEW  = (const float*)d_in[3];
    const int*   distEI = (const int*)  d_in[4];
    const float* distEW = (const float*)d_in[5];
    const int*   comEI  = (const int*)  d_in[6];
    const float* comEW  = (const float*)d_in[7];
    const float* W_rna1 = (const float*)d_in[8];
    const float* b_rna1 = (const float*)d_in[9];
    const float* W_rna2 = (const float*)d_in[10];
    const float* b_rna2 = (const float*)d_in[11];
    const float* W_p3   = (const float*)d_in[12];
    const float* b_p3   = (const float*)d_in[13];
    const float* W_sim  = (const float*)d_in[14];
    const float* b_sim  = (const float*)d_in[15];
    const float* W_dist = (const float*)d_in[16];
    const float* b_dist = (const float*)d_in[17];
    const float* W_f1   = (const float*)d_in[18];
    const float* b_f1   = (const float*)d_in[19];
    const float* W_f2   = (const float*)d_in[20];
    const float* b_f2   = (const float*)d_in[21];

    const int N = N_NODES, E = N_EDGES;
    const int SEC = N * OUT_F;
    float* out = (float*)d_out;
    float* out_xsim  = out;
    float* out_xdist = out + (size_t)SEC;
    float* out_fused = out + (size_t)2 * SEC;
    float* out_fpro  = out + (size_t)3 * SEC;
    float* out_pro   = out + (size_t)4 * SEC;

    // ---- workspace bump allocator (256B aligned) ----
    char* ws = (char*)d_ws;
    size_t off = 0;
    auto alloc = [&](size_t bytes) -> char* {
        char* p = ws + off;
        off = (off + bytes + 255) & ~(size_t)255;
        return p;
    };
    float*    dinvS  = (float*)alloc((size_t)N * 4);
    float*    dinvD  = (float*)alloc((size_t)N * 4);
    float*    dinvC  = (float*)alloc((size_t)N * 4);
    float*    normS  = (float*)alloc((size_t)E * 4);
    float*    normD  = (float*)alloc((size_t)E * 4);
    float*    normC  = (float*)alloc((size_t)E * 4);
    uint16_t* Xbf    = (uint16_t*)alloc((size_t)N * KP_RNA * 2);  // reused: hs|hd|hp (f32)
    uint16_t* W1t    = (uint16_t*)alloc((size_t)HID * KP_RNA * 2);
    uint16_t* W2t    = (uint16_t*)alloc((size_t)HID * KP_RNA * 2);
    float*    hbuf   = (float*)alloc((size_t)N * HID * 4);        // h1 then h2
    float*    aggbuf = (float*)alloc((size_t)N * HID * 4);        // aggS/aggD, later concat bufs
    uint16_t* xs_bf  = (uint16_t*)alloc((size_t)N * HID * 2);
    uint16_t* xd_bf  = (uint16_t*)alloc((size_t)N * HID * 2);
    uint16_t* WsimT  = (uint16_t*)alloc((size_t)OUT_F * HID * 2);
    uint16_t* WdistT = (uint16_t*)alloc((size_t)OUT_F * HID * 2);
    uint16_t* Wp3T   = (uint16_t*)alloc((size_t)OUT_F * KP_ADT * 2);
    uint16_t* Wf1T   = (uint16_t*)alloc((size_t)OUT_F * HID * 2);
    uint16_t* Wf2T   = (uint16_t*)alloc((size_t)OUT_F * HID * 2);
    uint16_t* adtBf  = (uint16_t*)alloc((size_t)N * KP_ADT * 2);
    (void)ws_size; (void)in_sizes; (void)n_in; (void)out_size;

    // region reuse (lifetimes are disjoint by launch order below)
    float*    hs        = (float*)Xbf;                 // N x 256
    float*    hd        = hs + (size_t)N * OUT_F;      // N x 256
    float*    hp        = hd + (size_t)N * OUT_F;      // N x 256 (61.4MB <= 80.6MB)
    uint16_t* concatBf  = (uint16_t*)aggbuf;           // N x 512 bf16
    uint16_t* concat2Bf = concatBf + (size_t)N * HID;  // N x 512 bf16 (40.96MB total)

    const int TB = 256;
    auto cdiv = [](long long a, long long b) { return (int)((a + b - 1) / b); };

    // ---- Phase A: degrees, dinv, per-edge norms (3 edge sets) ----
    const int*   EIs[3]  = { simEI, distEI, comEI };
    const float* EWs[3]  = { simEW, distEW, comEW };
    float*       DVs[3]  = { dinvS, dinvD, dinvC };
    float*       NMs[3]  = { normS, normD, normC };
    for (int s = 0; s < 3; ++s) {
        const int* srcp = EIs[s];
        const int* dstp = EIs[s] + E;
        k_fill1<<<cdiv(N, TB), TB, 0, stream>>>(DVs[s], N);
        k_scatter_deg<<<cdiv(E, TB), TB, 0, stream>>>(DVs[s], dstp, EWs[s], E);
        k_to_dinv<<<cdiv(N, TB), TB, 0, stream>>>(DVs[s], N);
        k_edge_norm<<<cdiv(E, TB), TB, 0, stream>>>(NMs[s], DVs[s], srcp, dstp, EWs[s], E);
    }

    // ---- Phase B: bf16 conversions ----
    k_cvt_pad_bf16<<<dim3(cdiv(KP_RNA, TB), N), TB, 0, stream>>>(Xbf, x_RNA, IN_C, KP_RNA);
    k_cvt_wT_bf16<<<dim3(cdiv(KP_RNA, TB), HID), TB, 0, stream>>>(W1t, W_rna1, IN_C, HID, KP_RNA);
    k_cvt_wT_bf16<<<dim3(cdiv(KP_RNA, TB), HID), TB, 0, stream>>>(W2t, W_rna2, IN_C, HID, KP_RNA);
    k_cvt_wT_bf16<<<dim3(cdiv(HID, TB), OUT_F), TB, 0, stream>>>(WsimT, W_sim, HID, OUT_F, HID);
    k_cvt_wT_bf16<<<dim3(cdiv(HID, TB), OUT_F), TB, 0, stream>>>(WdistT, W_dist, HID, OUT_F, HID);
    k_cvt_wT_bf16<<<dim3(cdiv(KP_ADT, TB), OUT_F), TB, 0, stream>>>(Wp3T, W_p3, QDIM, OUT_F, KP_ADT);
    k_cvt_wT_bf16<<<dim3(cdiv(HID, TB), OUT_F), TB, 0, stream>>>(Wf1T, W_f1, HID, OUT_F, HID);
    k_cvt_wT_bf16<<<dim3(cdiv(HID, TB), OUT_F), TB, 0, stream>>>(Wf2T, W_f2, HID, OUT_F, HID);
    k_cvt_pad_bf16<<<dim3(cdiv(KP_ADT, TB), N), TB, 0, stream>>>(adtBf, x_ADT, QDIM, KP_ADT);

    // GEMM launch geometry: wave = 32x32 patch, block = 4 waves = 32 rows x 128 cols
    const dim3 gemmBlk(32 * GEMM_WAVES);
    const dim3 gridHID(N / 32, HID / (32 * GEMM_WAVES));    // 625 x 4
    const dim3 gridOUT(N / 32, OUT_F / (32 * GEMM_WAVES));  // 625 x 2

    // ---- Phase C: conv1 (sim graph, relu) then conv2 (dist graph, relu) ----
    k_gemm_bf16_wmma<<<gridHID, gemmBlk, 0, stream>>>(Xbf, W1t, hbuf, nullptr, HID, KP_RNA);
    k_agg_init<<<cdiv((long long)N * HID, TB), TB, 0, stream>>>(aggbuf, hbuf, dinvS, b_rna1, N * HID, 9);
    k_agg_scatter<<<cdiv((long long)E * (HID / 4), TB), TB, 0, stream>>>(
        aggbuf, hbuf, simEI, simEI + E, normS, E * (HID / 4), HID, 7);
    k_relu_bf16<<<cdiv((long long)N * HID, TB), TB, 0, stream>>>(xs_bf, aggbuf, N * HID);

    k_gemm_bf16_wmma<<<gridHID, gemmBlk, 0, stream>>>(Xbf, W2t, hbuf, nullptr, HID, KP_RNA);
    k_agg_init<<<cdiv((long long)N * HID, TB), TB, 0, stream>>>(aggbuf, hbuf, dinvD, b_rna2, N * HID, 9);
    k_agg_scatter<<<cdiv((long long)E * (HID / 4), TB), TB, 0, stream>>>(
        aggbuf, hbuf, distEI, distEI + E, normD, E * (HID / 4), HID, 7);
    k_relu_bf16<<<cdiv((long long)N * HID, TB), TB, 0, stream>>>(xd_bf, aggbuf, N * HID);

    // ---- Phase D: x_sim / x_dist convolutions (Xbf region now hosts hs/hd/hp) ----
    k_gemm_bf16_wmma<<<gridOUT, gemmBlk, 0, stream>>>(xs_bf, WsimT, hs, nullptr, OUT_F, HID);
    k_gemm_bf16_wmma<<<gridOUT, gemmBlk, 0, stream>>>(xd_bf, WdistT, hd, nullptr, OUT_F, HID);
    k_agg_init<<<cdiv((long long)N * OUT_F, TB), TB, 0, stream>>>(out_xsim, hs, dinvS, b_sim, N * OUT_F, 8);
    k_agg_scatter<<<cdiv((long long)E * (OUT_F / 4), TB), TB, 0, stream>>>(
        out_xsim, hs, simEI, simEI + E, normS, E * (OUT_F / 4), OUT_F, 6);
    k_agg_init<<<cdiv((long long)N * OUT_F, TB), TB, 0, stream>>>(out_xdist, hd, dinvD, b_dist, N * OUT_F, 8);
    k_agg_scatter<<<cdiv((long long)E * (OUT_F / 4), TB), TB, 0, stream>>>(
        out_xdist, hd, distEI, distEI + E, normD, E * (OUT_F / 4), OUT_F, 6);

    // ---- Phase E: protein conv on common graph ----
    k_gemm_bf16_wmma<<<gridOUT, gemmBlk, 0, stream>>>(adtBf, Wp3T, hp, nullptr, OUT_F, KP_ADT);
    k_agg_init<<<cdiv((long long)N * OUT_F, TB), TB, 0, stream>>>(out_pro, hp, dinvC, b_p3, N * OUT_F, 8);
    k_agg_scatter<<<cdiv((long long)E * (OUT_F / 4), TB), TB, 0, stream>>>(
        out_pro, hp, comEI, comEI + E, normC, E * (OUT_F / 4), OUT_F, 6);

    // ---- Phase F: fused = [x_sim | x_dist] @ W_f1 + b_f1 ----
    k_pack_bf16<<<cdiv((long long)N * OUT_F, TB), TB, 0, stream>>>(concatBf, out_xsim, N * OUT_F, HID, 0, 8);
    k_pack_bf16<<<cdiv((long long)N * OUT_F, TB), TB, 0, stream>>>(concatBf, out_xdist, N * OUT_F, HID, OUT_F, 8);
    k_gemm_bf16_wmma<<<gridOUT, gemmBlk, 0, stream>>>(concatBf, Wf1T, out_fused, b_f1, OUT_F, HID);

    // ---- Phase G: fused_pro = [fused | pro] @ W_f2 + b_f2 ----
    k_pack_bf16<<<cdiv((long long)N * OUT_F, TB), TB, 0, stream>>>(concat2Bf, out_fused, N * OUT_F, HID, 0, 8);
    k_pack_bf16<<<cdiv((long long)N * OUT_F, TB), TB, 0, stream>>>(concat2Bf, out_pro, N * OUT_F, HID, OUT_F, 8);
    k_gemm_bf16_wmma<<<gridOUT, gemmBlk, 0, stream>>>(concat2Bf, Wf2T, out_fpro, b_f2, OUT_F, HID);
}
